// MHSA_75711683493974
// MI455X (gfx1250) — matrix operand order repaired and checked by reading, not compiled, not run on hardware
//
#include <hip/hip_runtime.h>
#include <hip/hip_bf16.h>

// ---------------------------------------------------------------------------
// MHSA forward for MI455X (gfx1250, wave32, WMMA).
// Shapes fixed by the reference: N=2, T=2048, D=1024, H=16, HW=64.
//
//   kernel 1: qkv_proj  -- y = x @ W^T for W in {Wq,Wk,Wv}, bf16 out in
//             [N,H,T,HW] layout in workspace (3 x 8 MB of d_ws).
//   kernel 2: flash_attn -- fused QK^T/8 + alibi + causal, online softmax,
//             PV; K tile staged with global_load_async_to_lds_b128.
// ---------------------------------------------------------------------------

#define NB   2
#define TT   2048
#define DD   1024
#define NH   16
#define HW   64
#define NEGB (-1e9f)

typedef __attribute__((ext_vector_type(16))) __bf16 v16bf;
typedef __attribute__((ext_vector_type(8)))  float  v8f;
typedef __attribute__((ext_vector_type(8)))  unsigned int v8u;

union FragAB { v8u u; v16bf b; };
union FragC  { v8f v; float f[8]; };

__device__ __forceinline__ v8f wmma_bf16(const unsigned int* a,
                                         const unsigned int* b, v8f c) {
  FragAB A, B;
#pragma unroll
  for (int i = 0; i < 8; ++i) { A.u[i] = a[i]; B.u[i] = b[i]; }
  return __builtin_amdgcn_wmma_f32_16x16x32_bf16(
      /*neg_a=*/false, A.b, /*neg_b=*/false, B.b,
      /*c_mod=*/(short)0, c, /*reuse_a=*/false, /*reuse_b=*/false);
}

// f32 -> bf16 (round to nearest even)
__device__ __forceinline__ unsigned short f2bf(float f) {
  unsigned int u = __builtin_bit_cast(unsigned int, f);
  u += 0x7FFFu + ((u >> 16) & 1u);
  return (unsigned short)(u >> 16);
}
__device__ __forceinline__ unsigned int pk_bf16(float a, float b) {
  return (unsigned int)f2bf(a) | ((unsigned int)f2bf(b) << 16);
}

// A-fragment (16x32 bf16): VGPR v, lanes 0-15 hold K=2v,2v+1;
// lanes 16-31 add +16.  Packed-dword index within a 16-dword row:
__device__ __forceinline__ int a_uidx(int v, int laneHalf) {
  return v + 8 * laneHalf;
}

// Async global->LDS copy of 16 bytes per lane (ASYNCcnt-tracked, gfx1250).
// ldsOff = addrspace(3) byte offset (low 32 bits of generic pointer).
__device__ __forceinline__ void async_b128_to_lds(unsigned int ldsOff,
                                                  const void* gaddr) {
  asm volatile("global_load_async_to_lds_b128 %0, %1, off"
               :: "v"(ldsOff), "v"(gaddr)
               : "memory");
}
__device__ __forceinline__ void wait_asynccnt0() {
  asm volatile("s_wait_asynccnt 0x0" ::: "memory");
}

// ---------------------------------------------------------------------------
// Kernel 1: y[m, nc] = sum_d x[m,d] * W[nc,d], written as bf16 [N,H,T,HW].
// block = 256 threads (8 waves, 4x2), tile 128(M) x 64(N), K step 32.
// ---------------------------------------------------------------------------
__global__ __launch_bounds__(256) void qkv_proj(
    const float* __restrict__ x,
    const float* __restrict__ Wq, const float* __restrict__ Wk,
    const float* __restrict__ Wv,
    unsigned short* __restrict__ Q, unsigned short* __restrict__ K,
    unsigned short* __restrict__ V) {
  __shared__ __align__(16) unsigned int Xsh[128 * 16]; // 128 x 32 bf16
  __shared__ __align__(16) unsigned int Wsh[64 * 16];  //  64 x 32 bf16

  const int tid      = threadIdx.x;
  const int wave     = tid >> 5;
  const int lane     = tid & 31;
  const int laneHalf = lane >> 4;
  const int l16      = lane & 15;

  const float* W = (blockIdx.z == 0) ? Wq : (blockIdx.z == 1) ? Wk : Wv;
  unsigned short* Out = (blockIdx.z == 0) ? Q : (blockIdx.z == 1) ? K : V;

  const int m0 = blockIdx.x * 128;
  const int n0 = blockIdx.y * 64;
  const int wm = (wave >> 1) * 32;  // wave M offset in tile
  const int wn = (wave & 1) * 32;   // wave N offset in tile

  float acc[2][2][8];
#pragma unroll
  for (int i = 0; i < 2; ++i)
#pragma unroll
    for (int j = 0; j < 2; ++j)
#pragma unroll
      for (int v = 0; v < 8; ++v) acc[i][j][v] = 0.0f;

  for (int k0 = 0; k0 < DD; k0 += 32) {
    __syncthreads();
    // stage x tile (128x32 f32 -> bf16)
    for (int i = tid; i < 128 * 8; i += 256) {
      int r = i >> 3, c4 = i & 7;
      const float* src = x + (size_t)(m0 + r) * DD + k0;
      float4 f = ((const float4*)src)[c4];
      if (k0 + 32 < DD) __builtin_prefetch(src + 32 + c4 * 4, 0, 1);
      Xsh[r * 16 + c4 * 2 + 0] = pk_bf16(f.x, f.y);
      Xsh[r * 16 + c4 * 2 + 1] = pk_bf16(f.z, f.w);
    }
    // stage W tile (64x32 f32 -> bf16); B element (k,n) = W[n0+n][k0+k]
    for (int i = tid; i < 64 * 8; i += 256) {
      int r = i >> 3, c4 = i & 7;
      const float* src = W + (size_t)(n0 + r) * DD + k0;
      float4 f = ((const float4*)src)[c4];
      if (k0 + 32 < DD) __builtin_prefetch(src + 32 + c4 * 4, 0, 1);
      Wsh[r * 16 + c4 * 2 + 0] = pk_bf16(f.x, f.y);
      Wsh[r * 16 + c4 * 2 + 1] = pk_bf16(f.z, f.w);
    }
    __syncthreads();

    unsigned int fa[2][8], fb[2][8];
#pragma unroll
    for (int v = 0; v < 8; ++v) {
      int ua = a_uidx(v, laneHalf);
      fa[0][v] = Xsh[(wm + l16) * 16 + ua];
      fa[1][v] = Xsh[(wm + 16 + l16) * 16 + ua];
      // B 32x16: VGPR v holds K=2v,2v+1 (+16 for upper half), col = l16
      int ub = v + 8 * laneHalf;
      fb[0][v] = Wsh[(wn + l16) * 16 + ub];
      fb[1][v] = Wsh[(wn + 16 + l16) * 16 + ub];
    }
#pragma unroll
    for (int i = 0; i < 2; ++i)
#pragma unroll
      for (int j = 0; j < 2; ++j) {
        FragC c;
#pragma unroll
        for (int v = 0; v < 8; ++v) c.f[v] = acc[i][j][v];
        c.v = wmma_bf16(fa[i], fb[j], c.v);
#pragma unroll
        for (int v = 0; v < 8; ++v) acc[i][j][v] = c.f[v];
      }
  }

  // epilogue: C/D layout row = v + 8*laneHalf, col = l16
#pragma unroll
  for (int i = 0; i < 2; ++i)
#pragma unroll
    for (int j = 0; j < 2; ++j)
#pragma unroll
      for (int v = 0; v < 8; ++v) {
        int m  = m0 + wm + i * 16 + v + 8 * laneHalf;
        int nc = n0 + wn + j * 16 + l16;
        int b = m >> 11, t = m & (TT - 1);
        int h = nc >> 6, c = nc & (HW - 1);
        Out[((size_t)(b * NH + h) * TT + t) * HW + c] = f2bf(acc[i][j][v]);
      }
}

// ---------------------------------------------------------------------------
// Kernel 2: flash attention. grid = (T/64, N*H), block = 128 (4 waves).
// Each wave: 16 q-rows, O accum 16x64 f32. k-blocks of 32.
// ---------------------------------------------------------------------------
__global__ __launch_bounds__(128) void flash_attn(
    const unsigned short* __restrict__ Qh, const unsigned short* __restrict__ Kh,
    const unsigned short* __restrict__ Vh, float* __restrict__ out) {
  __shared__ __align__(16) unsigned int   Ksh[32 * 32];       // 32 x 64 bf16
  __shared__ __align__(16) unsigned short VshT[64 * 32];      // transposed V
  __shared__ __align__(16) unsigned short Psh[4 * 16 * 32];   // per-wave P

  const int tid      = threadIdx.x;
  const int wave     = tid >> 5;
  const int lane     = tid & 31;
  const int laneHalf = lane >> 4;
  const int l16      = lane & 15;

  const int nh = blockIdx.y;
  const int bb = nh >> 4;          // batch
  const int h  = nh & (NH - 1);    // head
  const int qb = blockIdx.x;       // 64-row q block
  const int qw0 = qb * 64 + wave * 16;

  const float slope = exp2f(-0.5f * (float)(h + 1));  // 2^(-(h+1)*8/H)
  const size_t headBase = (size_t)(bb * NH + h) * TT * HW;

  // addrspace(3) byte offset of Ksh (low 32 bits of generic pointer)
  const unsigned int kshOff = (unsigned int)(size_t)(const void*)&Ksh[0];

  // Q A-fragments (two 16x32 chunks over HW), held in registers
  unsigned int fqa0[8], fqa1[8];
  {
    const unsigned int* Qrow =
        (const unsigned int*)(Qh + headBase + (size_t)(qw0 + l16) * HW);
#pragma unroll
    for (int v = 0; v < 8; ++v) {
      int ua = a_uidx(v, laneHalf);
      fqa0[v] = Qrow[ua];
      fqa1[v] = Qrow[16 + ua];
    }
  }

  float m[8], l[8], acc[4][8];
#pragma unroll
  for (int v = 0; v < 8; ++v) { m[v] = -3.0e38f; l[v] = 0.0f; }
#pragma unroll
  for (int j = 0; j < 4; ++j)
#pragma unroll
    for (int v = 0; v < 8; ++v) acc[j][v] = 0.0f;

  const int kend = qb * 64 + 64;  // causal bound for this block
  for (int kt = 0; kt < kend; kt += 32) {
    __syncthreads();
    // K tile: direct async global->LDS (row-major 32x64 bf16, contiguous 4KB)
    for (int i = tid; i < 256; i += 128) {
      const unsigned short* gk = Kh + headBase + (size_t)kt * HW + i * 8;
      async_b128_to_lds(kshOff + (unsigned int)(i * 16), gk);
      if (kt + 32 < kend) {
        __builtin_prefetch(gk + 32 * HW, 0, 1);                 // next K block
        __builtin_prefetch(Vh + headBase + (size_t)(kt + 32) * HW + i * 8, 0, 1);
      }
    }
    // V tile: through VGPRs with transpose so PV B-frags are single b32 reads
    for (int i = tid; i < 256; i += 128) {
      int r = i >> 3, c = i & 7;
      const uint4 dv =
          ((const uint4*)(Vh + headBase + (size_t)(kt + r) * HW))[c];
      unsigned int w[4] = {dv.x, dv.y, dv.z, dv.w};
#pragma unroll
      for (int q2 = 0; q2 < 4; ++q2) {
        int hw = c * 8 + q2 * 2;
        VshT[(hw + 0) * 32 + r] = (unsigned short)(w[q2] & 0xFFFFu);
        VshT[(hw + 1) * 32 + r] = (unsigned short)(w[q2] >> 16);
      }
    }
    wait_asynccnt0();   // K async copies from this wave have landed
    __syncthreads();    // all waves' K/V staging visible

    // S = Q K^T for two 16-col tiles (k positions kt+t*16 .. +15)
    float sblk[2][8];
#pragma unroll
    for (int t = 0; t < 2; ++t) {
      unsigned int fkb0[8], fkb1[8];
#pragma unroll
      for (int v = 0; v < 8; ++v) {
        int row = t * 16 + l16;             // k position (B column)
        int ub  = v + 8 * laneHalf;         // hw pair index
        fkb0[v] = Ksh[row * 32 + ub];       // hw 0..31
        fkb1[v] = Ksh[row * 32 + 16 + ub];  // hw 32..63
      }
      v8f sz = {};
      v8f st = wmma_bf16(fqa0, fkb0, sz);
      st = wmma_bf16(fqa1, fkb1, st);
      FragC su; su.v = st;
#pragma unroll
      for (int v = 0; v < 8; ++v) sblk[t][v] = su.f[v];
    }

    // scale + alibi + causal; online softmax update
#pragma unroll
    for (int t = 0; t < 2; ++t)
#pragma unroll
      for (int v = 0; v < 8; ++v) {
        int qpos = qw0 + v + 8 * laneHalf;
        int kpos = kt + t * 16 + l16;
        float s = sblk[t][v] * 0.125f + slope * (float)(kpos - qpos);
        if (kpos > qpos) s += NEGB;
        sblk[t][v] = s;
      }

#pragma unroll
    for (int v = 0; v < 8; ++v) {
      float rm = fmaxf(sblk[0][v], sblk[1][v]);
#pragma unroll
      for (int off = 1; off < 16; off <<= 1)
        rm = fmaxf(rm, __shfl_xor(rm, off, 32));
      float mn = fmaxf(m[v], rm);
      float alpha = __expf(m[v] - mn);
      float p0 = __expf(sblk[0][v] - mn);
      float p1 = __expf(sblk[1][v] - mn);
      sblk[0][v] = p0;
      sblk[1][v] = p1;
      float rs = p0 + p1;
#pragma unroll
      for (int off = 1; off < 16; off <<= 1)
        rs += __shfl_xor(rs, off, 32);
      l[v] = l[v] * alpha + rs;
      m[v] = mn;
#pragma unroll
      for (int j = 0; j < 4; ++j) acc[j][v] *= alpha;
    }

    // restage P (16x32) through per-wave LDS into A-fragment layout
#pragma unroll
    for (int t = 0; t < 2; ++t)
#pragma unroll
      for (int v = 0; v < 8; ++v)
        Psh[wave * 512 + (v + 8 * laneHalf) * 32 + t * 16 + l16] =
            f2bf(sblk[t][v]);

    unsigned int fpa[8], fvb[8];
    const unsigned int* P32 = (const unsigned int*)Psh;
#pragma unroll
    for (int v = 0; v < 8; ++v)
      fpa[v] = P32[wave * 256 + l16 * 16 + a_uidx(v, laneHalf)];

    // O += P * V   (4 column tiles of 16)
    const unsigned int* VT32 = (const unsigned int*)VshT;
#pragma unroll
    for (int j = 0; j < 4; ++j) {
#pragma unroll
      for (int v = 0; v < 8; ++v)
        fvb[v] = VT32[(j * 16 + l16) * 16 + v + 8 * laneHalf];
      FragC c;
#pragma unroll
      for (int v = 0; v < 8; ++v) c.f[v] = acc[j][v];
      c.v = wmma_bf16(fpa, fvb, c.v);
#pragma unroll
      for (int v = 0; v < 8; ++v) acc[j][v] = c.f[v];
    }
  }

  // normalize and store: out[b, qpos, h*64 + col]  (f32)
#pragma unroll
  for (int v = 0; v < 8; ++v) {
    int qpos = qw0 + v + 8 * laneHalf;
    float inv = 1.0f / l[v];
#pragma unroll
    for (int j = 0; j < 4; ++j)
      out[((size_t)bb * TT + qpos) * DD + h * HW + j * 16 + l16] =
          acc[j][v] * inv;
  }
}

// ---------------------------------------------------------------------------
extern "C" void kernel_launch(void* const* d_in, const int* in_sizes, int n_in,
                              void* d_out, int out_size, void* d_ws,
                              size_t ws_size, hipStream_t stream) {
  const float* x  = (const float*)d_in[0];
  // d_in[1] = additive causal mask: reproduced analytically in-kernel
  const float* Wq = (const float*)d_in[2];
  const float* Wk = (const float*)d_in[3];
  const float* Wv = (const float*)d_in[4];
  float* out = (float*)d_out;

  const size_t headElems = (size_t)NB * NH * TT * HW;  // 4 Mi elements
  unsigned short* Q = (unsigned short*)d_ws;
  unsigned short* K = Q + headElems;
  unsigned short* V = K + headElems;

  dim3 g1((NB * TT) / 128, DD / 64, 3);  // 32 x 16 x 3
  qkv_proj<<<g1, 256, 0, stream>>>(x, Wq, Wk, Wv, Q, K, V);

  dim3 g2(TT / 64, NB * NH);  // 32 x 32
  flash_attn<<<g2, 128, 0, stream>>>(Q, K, V, out);
}
